// STPAttention_82094004896537
// MI455X (gfx1250) — matrix operand order repaired and checked by reading, not compiled, not run on hardware
//
#include <hip/hip_runtime.h>
#include <hip/hip_bf16.h>
#include <math.h>

// ---------------------------------------------------------------------------
// Types for CDNA5 WMMA (wave32): v_wmma_f32_16x16x32_bf16
// ---------------------------------------------------------------------------
typedef __bf16 bf16_t;
typedef __attribute__((ext_vector_type(16))) __bf16 v16bf;
typedef __attribute__((ext_vector_type(8)))  __bf16 v8bf;
typedef __attribute__((ext_vector_type(8)))  float  v8f;

#define TSTRIDE 40   // LDS row stride (bf16): 80B = 5*16B aligned, staggers banks

// ---------------------------------------------------------------------------
// fp32 -> bf16 conversion
// ---------------------------------------------------------------------------
__global__ void cvt_f32_to_bf16(const float* __restrict__ in,
                                bf16_t* __restrict__ out, int n) {
    int i = blockIdx.x * blockDim.x + threadIdx.x;
    if (i < n) out[i] = (bf16_t)in[i];
}

// ---------------------------------------------------------------------------
// C[M][N] = A[M][K] @ B[N][K]^T   (A,B bf16 row-major, C f32 row-major)
//
// Block tile 128x128, K-tile 32, double-buffered LDS (40 KB).
// 256 thr = 8 waves arranged 4(M) x 2(N); wave tile 32x64 = 2x4 WMMA tiles.
// Per K-tile: 8 WMMAs / 12 ds_load_b128 / 1 barrier per wave.
// Global loads for tile t+1 issued before computing tile t (latency hidden);
// gfx1250 global_prefetch_b8 issued for tile t+2.
// ---------------------------------------------------------------------------
__global__ __launch_bounds__(256)
void gemm_bf16_nt_wmma(const bf16_t* __restrict__ A,
                       const bf16_t* __restrict__ B,
                       float* __restrict__ C,
                       int M, int N, int K) {
    __shared__ bf16_t lA[2][128 * TSTRIDE];
    __shared__ bf16_t lB[2][128 * TSTRIDE];

    const int tid   = threadIdx.x;
    const int lane  = tid & 31;
    const int wave  = tid >> 5;
    const int waveM = wave >> 1;        // 0..3 -> rows waveM*32
    const int waveN = wave & 1;         // 0..1 -> cols waveN*64
    const int half  = lane >> 4;        // lane group per ISA fragment layout
    const int r     = lane & 15;

    const int bm = blockIdx.y * 128;
    const int bn = blockIdx.x * 128;

    // cooperative load: thread t -> row t>>1 (0..127), 16 bf16 at col (t&1)*16
    const int  row = tid >> 1, col = (tid & 1) * 16;
    const bf16_t* gA = A + (long)(bm + row) * K + col;
    const bf16_t* gB = B + (long)(bn + row) * K + col;

    v8f acc[2][4];
    #pragma unroll
    for (int s = 0; s < 2; ++s)
        #pragma unroll
        for (int u = 0; u < 4; ++u) acc[s][u] = (v8f){};

    v8bf ra[2], rb[2];

    // ---- K-tile 0 -> LDS buffer 0 ----
    #pragma unroll
    for (int j = 0; j < 2; ++j) {
        ra[j] = *(const v8bf*)(gA + 8 * j);
        rb[j] = *(const v8bf*)(gB + 8 * j);
    }
    #pragma unroll
    for (int j = 0; j < 2; ++j) {
        *(v8bf*)&lA[0][row * TSTRIDE + col + 8 * j] = ra[j];
        *(v8bf*)&lB[0][row * TSTRIDE + col + 8 * j] = rb[j];
    }
    __syncthreads();

    const int ntiles = K >> 5;          // K / 32
    for (int t = 0; t < ntiles; ++t) {
        const int buf = t & 1;
        const int k0  = (t + 1) << 5;   // next tile's K offset

        if (t + 1 < ntiles) {
            if (t + 2 < ntiles) {       // gfx1250 global_prefetch_b8 (tile t+2)
                __builtin_prefetch(gA + k0 + 32, 0, 3);
                __builtin_prefetch(gB + k0 + 32, 0, 3);
            }
            #pragma unroll
            for (int j = 0; j < 2; ++j) {
                ra[j] = *(const v8bf*)(gA + k0 + 8 * j);
                rb[j] = *(const v8bf*)(gB + k0 + 8 * j);
            }
        }

        // ---- compute on LDS buffer `buf`: 2 A-frags, 4 B-frags, 8 WMMAs ----
        v16bf afrag[2];
        #pragma unroll
        for (int s = 0; s < 2; ++s) {
            // A frag: row = waveM*32 + s*16 + r
            //   elems 0..7  -> k =      8*half + j
            //   elems 8..15 -> k = 16 + 8*half + j
            const bf16_t* pa = &lA[buf][(waveM * 32 + s * 16 + r) * TSTRIDE + 8 * half];
            v8bf lo = *(const v8bf*)pa;
            v8bf hi = *(const v8bf*)(pa + 16);
            #pragma unroll
            for (int j = 0; j < 8; ++j) { afrag[s][j] = lo[j]; afrag[s][j + 8] = hi[j]; }
        }
        v16bf bfrag[4];
        #pragma unroll
        for (int u = 0; u < 4; ++u) {
            // B frag: n = waveN*64 + u*16 + r ; elems 0..7 -> k = 8*half + j, elems 8..15 -> k = 16 + 8*half + j
            const bf16_t* pb = &lB[buf][(waveN * 64 + u * 16 + r) * TSTRIDE + 8 * half];
            v8bf lo = *(const v8bf*)pb;
            v8bf hi = *(const v8bf*)(pb + 16);
            #pragma unroll
            for (int j = 0; j < 8; ++j) { bfrag[u][j] = lo[j]; bfrag[u][j + 8] = hi[j]; }
        }
        #pragma unroll
        for (int s = 0; s < 2; ++s)
            #pragma unroll
            for (int u = 0; u < 4; ++u)
                acc[s][u] = __builtin_amdgcn_wmma_f32_16x16x32_bf16(
                    false, afrag[s], false, bfrag[u], (short)0, acc[s][u],
                    false, false);

        // ---- stage tile t+1 into the other LDS buffer ----
        if (t + 1 < ntiles) {
            #pragma unroll
            for (int j = 0; j < 2; ++j) {
                *(v8bf*)&lA[buf ^ 1][row * TSTRIDE + col + 8 * j] = ra[j];
                *(v8bf*)&lB[buf ^ 1][row * TSTRIDE + col + 8 * j] = rb[j];
            }
            __syncthreads();
        }
    }

    // D layout: n = lane&15 ; VGPR v -> row v + 8*half
    #pragma unroll
    for (int s = 0; s < 2; ++s) {
        #pragma unroll
        for (int u = 0; u < 4; ++u) {
            const int cn = bn + waveN * 64 + u * 16 + r;
            #pragma unroll
            for (int v = 0; v < 8; ++v) {
                const int cm = bm + waveM * 32 + s * 16 + 8 * half + v;
                C[(long)cm * N + cn] = acc[s][u][v];
            }
        }
    }
}

// ---------------------------------------------------------------------------
// Per-head constant matrices: retention & coef  (H*dk*dk = 65536 elems)
// ---------------------------------------------------------------------------
__global__ void head_consts(const float* __restrict__ W_LTM,
                            const float* __restrict__ V_gs,
                            const float* __restrict__ V_T0,
                            const float* __restrict__ beta_tau,
                            const float* __restrict__ beta_gm,
                            const float* __restrict__ C_ch,
                            const float* __restrict__ gamma,
                            const float* __restrict__ alpha_ppd,
                            const float* __restrict__ IC_threshold,
                            float* __restrict__ retention,
                            float* __restrict__ coef) {
    int i = blockIdx.x * blockDim.x + threadIdx.x;   // over H*64*64
    int h = i >> 12;
    float vn  = V_gs[h] - V_T0[h] + W_LTM[i];        // U_T = 1
    float sp  = (vn > 20.f) ? vn : log1pf(expf(vn)); // softplus
    float gch = beta_tau[h] * sp;
    float sig = 1.f / (1.f + expf(-vn));
    float G   = beta_gm[h] * sp * sig;
    float sm  = tanhf(alpha_ppd[0] * (gch - IC_threshold[0]));
    retention[i] = expf(-gch / C_ch[h]);
    coef[i]      = gamma[h] * sm * G;
}

// ---------------------------------------------------------------------------
// Sequential scan: one block per (b,h). Thread t owns F[d][e0..e0+16) in
// registers (d=t>>2, e0=(t&3)*16). Double-buffered LDS staging of q/k/v
// -> ONE barrier per timestep; next step's q/k/v prefetched into registers;
// LayerNorm stats computed redundantly per-wave with shuffles (no barrier).
// ---------------------------------------------------------------------------
__global__ __launch_bounds__(256)
void stp_scan(const float* __restrict__ qkv,    // [B, L, 3*D]
              const float* __restrict__ W_LTM,  // [H, dk, dk]
              const float* __restrict__ retention,
              const float* __restrict__ coef,
              float* __restrict__ y,            // [B, L, D]
              int L) {
    const int D = 1024, H = 16, dk = 64;
    const int b = blockIdx.x / H;
    const int h = blockIdx.x % H;
    const int t    = threadIdx.x;
    const int lane = t & 31;
    const int d    = t >> 2;
    const int e0   = (t & 3) * 16;

    float F[16], Wr[16], Rr[16], Cr[16];
    const long base = (long)h * dk * dk + (long)d * dk + e0;
    #pragma unroll
    for (int j = 0; j < 16; ++j) {
        F[j]  = 0.f;
        Wr[j] = W_LTM[base + j];
        Rr[j] = retention[base + j];
        Cr[j] = coef[base + j];
    }

    // sbuf[buf][0..63]=q, [64..127]=k(raw), [128..191]=v
    __shared__ float sbuf[2][192];

    const long row_stride = 3 * D;
    const float* pbase = qkv + (long)b * L * row_stride + h * dk;
    float*       y_b   = y   + (long)b * L * D;

    // thread t<192 stages one float per step: which = t>>6 (q/k/v), idx = t&63
    const long goff = (long)(t >> 6) * D + (t & 63);

    float nxt = 0.f;
    if (t < 192) nxt = pbase[goff];                       // step 0

    for (int l = 0; l < L; ++l) {
        float* sb = sbuf[l & 1];
        if (t < 192) sb[t] = nxt;
        __syncthreads();                                  // the only barrier

        // prefetch next step's q/k/v (latency hidden behind compute)
        if (t < 192 && l + 1 < L)
            nxt = pbase[(long)(l + 1) * row_stride + goff];

        // LayerNorm stats over dk=64, redundantly per wave (shuffle reduce)
        float a = sb[64 + lane], c = sb[96 + lane];
        float s = a + c, ss = a * a + c * c;
        #pragma unroll
        for (int o = 16; o > 0; o >>= 1) {
            s  += __shfl_down(s, o, 32);
            ss += __shfl_down(ss, o, 32);
        }
        s  = __shfl(s, 0, 32);
        ss = __shfl(ss, 0, 32);
        const float mu   = s * (1.f / 64.f);
        const float var  = ss * (1.f / 64.f) - mu * mu;
        const float rstd = rsqrtf(var + 1e-5f);

        const float vd = sb[128 + d];
        float qsum = 0.f;
        #pragma unroll
        for (int j = 0; j < 16; ++j) {
            float kn = (sb[64 + e0 + j] - mu) * rstd;
            float Fn = Rr[j] * F[j] + Cr[j] * (vd * kn);
            F[j] = Fn;
            qsum += (Wr[j] + Fn) * sb[e0 + j];
        }
        // reduce 4 partials of row d (threads 4d..4d+3 are in the same wave)
        qsum += __shfl_xor(qsum, 1, 32);
        qsum += __shfl_xor(qsum, 2, 32);
        if ((t & 3) == 0) y_b[(long)l * D + h * dk + d] = qsum;
        // no trailing barrier: next step writes the other LDS buffer
    }
}

// ---------------------------------------------------------------------------
// Launch
// ---------------------------------------------------------------------------
extern "C" void kernel_launch(void* const* d_in, const int* in_sizes, int n_in,
                              void* d_out, int out_size, void* d_ws, size_t ws_size,
                              hipStream_t stream) {
    const int B = 2, L = 2048, D = 1024, H = 16;
    const int M = B * L;               // 4096
    const int K = D;                   // 1024
    const int N_qkv = 3 * D;           // 3072

    const float* x      = (const float*)d_in[0];
    const float* W_qkv  = (const float*)d_in[1];
    const float* W_o    = (const float*)d_in[2];
    const float* W_LTM  = (const float*)d_in[3];
    const float* V_gs   = (const float*)d_in[4];
    const float* V_T0   = (const float*)d_in[5];
    const float* b_tau  = (const float*)d_in[6];
    const float* b_gm   = (const float*)d_in[7];
    const float* C_ch   = (const float*)d_in[8];
    const float* gam    = (const float*)d_in[9];
    const float* a_ppd  = (const float*)d_in[10];
    const float* IC_th  = (const float*)d_in[11];

    char* ws = (char*)d_ws;
    auto carve = [&](size_t bytes) {
        char* p = ws;
        ws += (bytes + 255) & ~(size_t)255;
        return p;
    };
    bf16_t* xb   = (bf16_t*)carve((size_t)M * K * sizeof(bf16_t));
    bf16_t* wqb  = (bf16_t*)carve((size_t)N_qkv * K * sizeof(bf16_t));
    bf16_t* wob  = (bf16_t*)carve((size_t)D * D * sizeof(bf16_t));
    float*  qkv  = (float*) carve((size_t)M * N_qkv * sizeof(float));
    float*  ret  = (float*) carve((size_t)H * 64 * 64 * sizeof(float));
    float*  coef = (float*) carve((size_t)H * 64 * 64 * sizeof(float));
    float*  yf   = (float*) carve((size_t)M * D * sizeof(float));
    bf16_t* yb   = (bf16_t*)carve((size_t)M * D * sizeof(bf16_t));

    // 1) fp32 -> bf16 for the GEMM operands
    int nx = M * K, nwq = N_qkv * K, nwo = D * D;
    cvt_f32_to_bf16<<<(nx  + 255) / 256, 256, 0, stream>>>(x,     xb,  nx);
    cvt_f32_to_bf16<<<(nwq + 255) / 256, 256, 0, stream>>>(W_qkv, wqb, nwq);
    cvt_f32_to_bf16<<<(nwo + 255) / 256, 256, 0, stream>>>(W_o,   wob, nwo);

    // 2) qkv = x @ W_qkv^T   (4096 x 3072 x 1024)
    gemm_bf16_nt_wmma<<<dim3(N_qkv / 128, M / 128), 256, 0, stream>>>(
        xb, wqb, qkv, M, N_qkv, K);

    // 3) per-head retention / coef matrices
    head_consts<<<(H * 64 * 64) / 256, 256, 0, stream>>>(
        W_LTM, V_gs, V_T0, b_tau, b_gm, C_ch, gam, a_ppd, IC_th, ret, coef);

    // 4) the serial recurrence (32 independent chains)
    stp_scan<<<B * H, 256, 0, stream>>>(qkv, W_LTM, ret, coef, yf, L);

    // 5) y -> bf16, then out = y @ W_o^T  (4096 x 1024 x 1024)
    int ny = M * D;
    cvt_f32_to_bf16<<<(ny + 255) / 256, 256, 0, stream>>>(yf, yb, ny);
    gemm_bf16_nt_wmma<<<dim3(D / 128, M / 128), 256, 0, stream>>>(
        yb, wob, (float*)d_out, M, D, K);
}
